// slice_49151605735608
// MI455X (gfx1250) — compile-verified, matrix operand order checked by reading, not run on hardware
//
#include <hip/hip_runtime.h>
#include <hip/hip_bf16.h>

typedef __attribute__((ext_vector_type(16))) _Float16 v16h;
typedef __attribute__((ext_vector_type(8)))  float    v8f;
typedef __attribute__((ext_vector_type(2)))  float    v2f;

#define BATCH 8
#define C     32
#define HW    512
#define TS    64
#define L     4096
#define NCH   8
#define NBK   512   // BATCH * 64 tiles
#define SCW   66    // LDS image width: 64 cols + 1 zero halo each side

// For k in 0..63: (si,sj,mask) — last (i,j) with i*j==k+1 under i-major loop
// order == largest valid i.
__device__ __forceinline__ void slice_map(int k, int& si, int& sj, int& m) {
  si = 0; sj = 0; m = 0;
  const int v = k + 1;
  for (int i = 8; i >= 1; --i) {
    if (v % i == 0 && v / i <= 8) { si = i - 1; sj = v / i - 1; m = 1; return; }
  }
}

// ---------------------------------------------------------------------------
// Kernel 1: per-tile 3x3 conv (SAME, zero pad per 64x64 slice) + bias + relu.
// Implicit GEMM: 9 taps, each one v_wmma_f32_16x16x32_f16 k-step (K = 32 ch).
// LDS image has zero halo columns so fragment loads are branch-free (no
// exec-mask churn between WMMAs).
// ---------------------------------------------------------------------------
__global__ void __launch_bounds__(256)
k_conv3x3_wmma(const float* __restrict__ x, const float* __restrict__ w30,
               const float* __restrict__ b30, float* __restrict__ Bout) {
  __shared__ _Float16 smem[10 * SCW * C];   // ~42 KB, layout [row][col][c]
  const int bk = blockIdx.x;
  const int b = bk >> 6, k = bk & 63;
  int si, sj, mk; slice_map(k, si, sj, mk);
  const int tid = threadIdx.x;
  const int lane = tid & 31, wave = tid >> 5;

  // Zero the halo columns (col 0 and col 65) once; strip loads never touch them.
  for (int t = tid; t < 2 * 10 * C; t += 256) {
    const int r = t >> 6;                 // /64
    const int e = t & 63;
    const int col = (e < 32) ? 0 : (SCW - 1);
    const int c = e & 31;
    smem[(r * SCW + col) * C + c] = (_Float16)0.0f;
  }

  // A fragments (weights, f16): M=out-ch (rows 8..15 zero), K=in-ch.
  // ISA 16-bit A layout: lane&15 = M; element e -> K=(e<8?e:e+8)+8*(lane>=16).
  v16h afrag[9];
  {
    const int m  = lane & 15;
    const int c0 = (lane >= 16) ? 8 : 0;
    for (int kh = 0; kh < 3; ++kh)
      for (int kw = 0; kw < 3; ++kw) {
        v16h a;
        for (int e = 0; e < 16; ++e) {
          const int c = ((e < 8) ? e : (e + 8)) + c0;
          const float wv = (m < NCH) ? w30[((m * C + c) * 3 + kh) * 3 + kw] : 0.0f;
          a[e] = (_Float16)wv;
        }
        afrag[kh * 3 + kw] = a;
      }
  }
  const float* xb = x + (size_t)b * C * HW * HW;

  for (int strip = 0; strip < 8; ++strip) {
    const int ys = strip * 8;
    // Strip loader: float4 global loads (16B aligned), 4 b16 LDS stores each.
    // Items: 10 rows x 32 ch x 16 col-groups = 5120.
    for (int t = tid; t < 10 * C * 16; t += 256) {
      const int r   = t >> 9;             // / (32*16)
      const int rem = t & 511;
      const int c   = rem >> 4;
      const int col = (rem & 15) << 2;
      const int yimg = ys - 1 + r;
      float4 v = make_float4(0.0f, 0.0f, 0.0f, 0.0f);
      if (mk && yimg >= 0 && yimg < TS) {
        const float* gp =
            &xb[((size_t)c * HW + (si * TS + yimg)) * HW + (sj * TS + col)];
        v = *(const float4*)gp;
        if (yimg + 8 < TS)  // hint next strip into cache (global_prefetch_b8)
          __builtin_prefetch(gp + (size_t)8 * HW, 0, 1);
      }
      _Float16* sp = &smem[(r * SCW + col + 1) * C + c];
      sp[0 * C] = (_Float16)v.x;
      sp[1 * C] = (_Float16)v.y;
      sp[2 * C] = (_Float16)v.z;
      sp[3 * C] = (_Float16)v.w;
    }
    __syncthreads();

    const int row  = wave;               // 8 waves, one output row each
    const int npix = lane & 15;
    const int c0   = (lane >= 16) ? 8 : 0;
    for (int xc = 0; xc < 4; ++xc) {
      v8f acc = {};
      for (int kh = 0; kh < 3; ++kh) {
        const int rin = row + kh;
        for (int kw = 0; kw < 3; ++kw) {
          // halo-padded column index: ximg+1 = xc*16 + npix + kw, in [0,65]
          const _Float16* p =
              &smem[(rin * SCW + xc * 16 + npix + kw) * C + c0];
          v16h bfrag;
          for (int e = 0; e < 8; ++e) bfrag[e]     = p[e];
          for (int e = 0; e < 8; ++e) bfrag[e + 8] = p[e + 16];
          acc = __builtin_amdgcn_wmma_f32_16x16x32_f16(
              false, afrag[kh * 3 + kw], false, bfrag, (short)0, acc, false, false);
        }
      }
      if (lane < 16) {  // lanes 0..15 hold M=0..7 (valid out channels)
        const int l = (ys + row) * TS + xc * 16 + lane;
        float* outp = Bout + (size_t)bk * NCH * L + l;
        for (int r = 0; r < NCH; ++r) {
          float v = acc[r] + b30[r];
          outp[(size_t)r * L] = v > 0.0f ? v : 0.0f;
        }
      }
    }
    __syncthreads();
  }
}

// ---------------------------------------------------------------------------
// Kernel 2: fs2[bk,n,c] = sum_l B[bk,n,l] * fs[bk,c,l]   (K = 4096)
// ---------------------------------------------------------------------------
__global__ void __launch_bounds__(256)
k_fs2(const float* __restrict__ x, const float* __restrict__ Bin,
      float* __restrict__ fs2) {
  __shared__ float sB[NCH * TS];
  __shared__ float sF[C * TS];
  const int bk = blockIdx.x;
  const int b = bk >> 6, k = bk & 63;
  int si, sj, mk; slice_map(k, si, sj, mk);
  const int tid = threadIdx.x;
  const int n = tid >> 5, c = tid & 31;
  const float* xb = x + (size_t)b * C * HW * HW;
  const float* Bp = Bin + (size_t)bk * NCH * L;
  float acc = 0.0f;
  for (int y = 0; y < TS; ++y) {
    for (int t = tid; t < NCH * TS; t += 256)
      sB[t] = Bp[(size_t)(t >> 6) * L + y * TS + (t & 63)];
    for (int t = tid; t < C * TS; t += 256) {
      const int cc = t >> 6, col = t & 63;
      sF[t] = mk ? xb[((size_t)cc * HW + si * TS + y) * HW + sj * TS + col] : 0.0f;
    }
    __syncthreads();
    for (int li = 0; li < TS; ++li) acc += sB[n * TS + li] * sF[c * TS + li];
    __syncthreads();
  }
  fs2[(size_t)bk * (NCH * C) + n * C + c] = acc;
}

// ---------------------------------------------------------------------------
// Kernel 3: fs3 = relu(w11 @ fs2 + b11); fs5[o,n] = relu(sum_c w10[o,c]*fs3[n,c]+b10)
// ---------------------------------------------------------------------------
__global__ void __launch_bounds__(256)
k_fs5(const float* __restrict__ fs2, const float* __restrict__ w11,
      const float* __restrict__ b11, const float* __restrict__ w10,
      const float* __restrict__ b10, float* __restrict__ fs5) {
  __shared__ float s2[NCH * C];
  __shared__ float s3[NCH * C];
  const int bk = blockIdx.x, tid = threadIdx.x;
  s2[tid] = fs2[(size_t)bk * (NCH * C) + tid];
  __syncthreads();
  {
    const int o = tid >> 5, c = tid & 31;
    float s = b11[o];
    for (int n = 0; n < NCH; ++n) s += w11[o * NCH + n] * s2[n * C + c];
    s3[o * C + c] = s > 0.0f ? s : 0.0f;
  }
  __syncthreads();
  {
    const int o = tid >> 3, n = tid & 7;
    float s = b10[o];
    for (int c = 0; c < C; ++c) s += w10[o * C + c] * s3[n * C + c];
    fs5[(size_t)bk * (C * NCH) + o * NCH + n] = s > 0.0f ? s : 0.0f;
  }
}

// ---------------------------------------------------------------------------
// Kernel 4: fg[b,c,ij] = mean of 64x64 tile of x
// ---------------------------------------------------------------------------
__global__ void __launch_bounds__(128)
k_fg(const float* __restrict__ x, float* __restrict__ fg) {
  const int id = blockIdx.x;
  const int b = id >> 11, c = (id >> 6) & 31, ij = id & 63;
  const int i = ij >> 3, j = ij & 7;
  const float* p = x + ((size_t)(b * C + c) * HW + i * TS) * HW + j * TS;
  float s = 0.0f;
  for (int t = threadIdx.x; t < L; t += 128)
    s += p[(size_t)(t >> 6) * HW + (t & 63)];
  __shared__ float red[128];
  red[threadIdx.x] = s;
  __syncthreads();
  for (int off = 64; off > 0; off >>= 1) {
    if (threadIdx.x < off) red[threadIdx.x] += red[threadIdx.x + off];
    __syncthreads();
  }
  if (threadIdx.x == 0) fg[id] = red[0] * (1.0f / (float)L);
}

// ---------------------------------------------------------------------------
// Kernel 5: gate MLP  g[b,k,c] = sigmoid(w12c @ relu(w12b @ relu(w12a @ fg1)))
// ---------------------------------------------------------------------------
__global__ void __launch_bounds__(256)
k_gate(const float* __restrict__ fg, const float* __restrict__ w12a,
       const float* __restrict__ b12a, const float* __restrict__ w12b,
       const float* __restrict__ b12b, const float* __restrict__ w12c,
       const float* __restrict__ b12c, float* __restrict__ g) {
  __shared__ float f1[64 * C];
  __shared__ float g1[NCH * C];
  __shared__ float g2[NCH * C];
  const int b = blockIdx.x, tid = threadIdx.x;
  for (int t = tid; t < 64 * C; t += 256) {
    const int ij = t >> 5, c = t & 31;
    f1[t] = fg[(size_t)b * 2048 + c * 64 + ij];
  }
  __syncthreads();
  { const int o = tid >> 5, c = tid & 31;
    float s = b12a[o];
    for (int kk = 0; kk < 64; ++kk) s += w12a[o * 64 + kk] * f1[kk * C + c];
    g1[o * C + c] = s > 0.0f ? s : 0.0f; }
  __syncthreads();
  { const int o = tid >> 5, c = tid & 31;
    float s = b12b[o];
    for (int n = 0; n < NCH; ++n) s += w12b[o * NCH + n] * g1[n * C + c];
    g2[o * C + c] = s > 0.0f ? s : 0.0f; }
  __syncthreads();
  for (int t = tid; t < 64 * C; t += 256) {
    const int kk = t >> 5, c = t & 31;
    float s = b12c[kk];
    for (int n = 0; n < NCH; ++n) s += w12c[kk * NCH + n] * g2[n * C + c];
    g[(size_t)b * 2048 + t] = 1.0f / (1.0f + __expf(-s));
  }
}

// ---------------------------------------------------------------------------
// Kernel 6: Wn[bk][o][n] = sum_c w31[o,c] * g[b,k,c] * fs5[bk,c,n]
// (collapses fs6 -> gate -> gather -> 1x1 conv into one 32x8 matrix per tile)
// ---------------------------------------------------------------------------
__global__ void __launch_bounds__(256)
k_wn(const float* __restrict__ w31, const float* __restrict__ g,
     const float* __restrict__ fs5, float* __restrict__ Wn) {
  const int bk = blockIdx.x;
  const int b = bk >> 6, k = bk & 63;
  const int tid = threadIdx.x;
  const int o = tid >> 3, n = tid & 7;
  const float* gp = g + (size_t)b * 2048 + k * C;
  const float* fp = fs5 + (size_t)bk * (C * NCH);
  float s = 0.0f;
  for (int c = 0; c < C; ++c) s += w31[o * C + c] * gp[c] * fp[c * NCH + n];
  Wn[(size_t)bk * 256 + tid] = s;
}

// ---------------------------------------------------------------------------
// Kernel 7: tail GEMM D[32,4096] = Wn[32,8] x B[8,4096] via v_wmma_f32_16x16x4_f32,
// fused with bias/BN/relu/residual. One block per output tile (b,i,j).
// ---------------------------------------------------------------------------
__global__ void __launch_bounds__(256)
k_tail_wmma(const float* __restrict__ x, const float* __restrict__ Bin,
            const float* __restrict__ Wn, const float* __restrict__ b31,
            const float* __restrict__ gam, const float* __restrict__ bet,
            const float* __restrict__ mean, const float* __restrict__ var,
            float* __restrict__ y) {
  const int id = blockIdx.x;
  const int b = id >> 6, ij = id & 63;
  const int i = ij >> 3, j = ij & 7;
  const int k = (i + 1) * (j + 1) - 1;     // uns[i][j]
  const int bk = b * 64 + k;
  const int tid = threadIdx.x, lane = tid & 31, wave = tid >> 5;
  const float* Wp = Wn + (size_t)bk * 256;

  // A fragments (f32 16x16x4): lane&15 = M; element e -> K = e + 2*(lane>=16)
  const int m     = lane & 15;
  const int kbase = (lane >= 16) ? 2 : 0;
  const int mo    = (lane >= 16) ? 8 : 0;
  v2f a[2][2];
  for (int t = 0; t < 2; ++t)
    for (int ks = 0; ks < 2; ++ks) {
      a[t][ks][0] = Wp[(t * 16 + m) * NCH + ks * 4 + kbase + 0];
      a[t][ks][1] = Wp[(t * 16 + m) * NCH + ks * 4 + kbase + 1];
    }
  float sc[2][8], sh[2][8];
  for (int t = 0; t < 2; ++t)
    for (int r = 0; r < 8; ++r) {
      const int o = t * 16 + r + mo;
      const float s = gam[o] * rsqrtf(var[o] + 1e-5f);
      sc[t][r] = s;
      sh[t][r] = (b31[o] - mean[o]) * s + bet[o];
    }

  const float* Bp = Bin + (size_t)bk * NCH * L;
  const size_t xoff = ((size_t)(b * C) * HW + i * TS) * HW + j * TS;
  const int npix = lane & 15;

  for (int q = 0; q < 32; ++q) {           // 256 N-tiles / 8 waves
    const int l0 = (wave * 32 + q) * 16;
    v2f bf0, bf1;
    bf0[0] = Bp[(size_t)(0 + kbase) * L + l0 + npix];
    bf0[1] = Bp[(size_t)(1 + kbase) * L + l0 + npix];
    bf1[0] = Bp[(size_t)(4 + kbase) * L + l0 + npix];
    bf1[1] = Bp[(size_t)(5 + kbase) * L + l0 + npix];
    v8f d0 = {}, d1 = {};
    d0 = __builtin_amdgcn_wmma_f32_16x16x4_f32(false, a[0][0], false, bf0, (short)0, d0, false, false);
    d0 = __builtin_amdgcn_wmma_f32_16x16x4_f32(false, a[0][1], false, bf1, (short)0, d0, false, false);
    d1 = __builtin_amdgcn_wmma_f32_16x16x4_f32(false, a[1][0], false, bf0, (short)0, d1, false, false);
    d1 = __builtin_amdgcn_wmma_f32_16x16x4_f32(false, a[1][1], false, bf1, (short)0, d1, false, false);

    const int l = l0 + npix;
    const size_t base = xoff + (size_t)(l >> 6) * HW + (l & 63);
    for (int r = 0; r < 8; ++r) {
      { const int o = r + mo;
        float v = d0[r] * sc[0][r] + sh[0][r];
        v = v > 0.0f ? v : 0.0f;
        const size_t addr = base + (size_t)o * HW * HW;
        y[addr] = v + x[addr]; }
      { const int o = 16 + r + mo;
        float v = d1[r] * sc[1][r] + sh[1][r];
        v = v > 0.0f ? v : 0.0f;
        const size_t addr = base + (size_t)o * HW * HW;
        y[addr] = v + x[addr]; }
    }
  }
}

extern "C" void kernel_launch(void* const* d_in, const int* in_sizes, int n_in,
                              void* d_out, int out_size, void* d_ws, size_t ws_size,
                              hipStream_t stream) {
  const float* x    = (const float*)d_in[0];
  const float* w30  = (const float*)d_in[1];
  const float* b30  = (const float*)d_in[2];
  const float* w10  = (const float*)d_in[3];
  const float* b10  = (const float*)d_in[4];
  const float* w11  = (const float*)d_in[5];
  const float* b11  = (const float*)d_in[6];
  const float* w12a = (const float*)d_in[7];
  const float* b12a = (const float*)d_in[8];
  const float* w12b = (const float*)d_in[9];
  const float* b12b = (const float*)d_in[10];
  const float* w12c = (const float*)d_in[11];
  const float* b12c = (const float*)d_in[12];
  const float* w31  = (const float*)d_in[13];
  const float* b31  = (const float*)d_in[14];
  const float* gam  = (const float*)d_in[15];
  const float* bet  = (const float*)d_in[16];
  const float* mean = (const float*)d_in[17];
  const float* var  = (const float*)d_in[18];
  float* y  = (float*)d_out;
  float* ws = (float*)d_ws;

  float* wsB   = ws;                              // 512*8*4096  (64 MB)
  float* wsFs2 = wsB   + (size_t)NBK * NCH * L;   // 512*256
  float* wsFs5 = wsFs2 + (size_t)NBK * 256;       // 512*256
  float* wsFg  = wsFs5 + (size_t)NBK * 256;       // 8*2048
  float* wsG   = wsFg  + (size_t)8 * 2048;        // 8*2048
  float* wsWn  = wsG   + (size_t)8 * 2048;        // 512*256

  k_conv3x3_wmma<<<NBK, 256, 0, stream>>>(x, w30, b30, wsB);
  k_fs2<<<NBK, 256, 0, stream>>>(x, wsB, wsFs2);
  k_fs5<<<NBK, 256, 0, stream>>>(wsFs2, w11, b11, w10, b10, wsFs5);
  k_fg<<<BATCH * C * 64, 128, 0, stream>>>(x, wsFg);
  k_gate<<<BATCH, 256, 0, stream>>>(wsFg, w12a, b12a, w12b, b12b, w12c, b12c, wsG);
  k_wn<<<NBK, 256, 0, stream>>>(w31, wsG, wsFs5, wsWn);
  k_tail_wmma<<<NBK, 256, 0, stream>>>(x, wsB, wsWn, b31, gam, bet, mean, var, y);
}
